// Model_55637006352648
// MI455X (gfx1250) — compile-verified, matrix operand order checked by reading, not run on hardware
//
#include <hip/hip_runtime.h>
#include <hip/hip_bf16.h>

typedef __attribute__((ext_vector_type(16))) _Float16 v16h;
typedef __attribute__((ext_vector_type(8)))  _Float16 v8h;
typedef __attribute__((ext_vector_type(8)))  float    v8f;

#define PI_F 3.14159265358979f

// ---------------------------------------------------------------------------
// WMMA helpers (CDNA5 wave32, V_WMMA_F32_16X16X32_F16)
// ---------------------------------------------------------------------------
__device__ __forceinline__ v8f wmma16(v16h a, v16h b, v8f c) {
  return __builtin_amdgcn_wmma_f32_16x16x32_f16(false, a, false, b,
                                                (short)0, c, false, false);
}

// A-layout fragment: element (row, k) at src[row*stride + k] (contiguous in k).
// Lane L (0..15): row = row0+L, kb = k0; lanes 16..31: row0+L-16, kb = k0+8.
// halves 0..7 -> K = kb..kb+7 ; halves 8..15 -> K = kb+16..kb+23.
// stride must be a multiple of 8 (16B alignment for the v8h loads).
__device__ __forceinline__ v16h frag_row(const _Float16* src, int stride,
                                         int row0, int k0) {
  int lane = threadIdx.x & 31;
  int r = row0 + (lane & 15);
  int kb = k0 + ((lane >> 4) << 3);
  const _Float16* p = src + r * stride + kb;
  v8h lo = *(const v8h*)(p);
  v8h hi = *(const v8h*)(p + 16);
  v16h o;
#pragma unroll
  for (int i = 0; i < 8; i++) { o[i] = lo[i]; o[i + 8] = hi[i]; }
  return o;
}

// B-layout (mirror of A) via scalar gather: element (k, n) at src[k*stride + n].
__device__ __forceinline__ v16h frag_col(const _Float16* src, int stride,
                                         int n0, int k0) {
  int lane = threadIdx.x & 31;
  int n = n0 + (lane & 15);
  int kb = k0 + ((lane >> 4) << 3);
  v16h o;
#pragma unroll
  for (int i = 0; i < 8; i++) {
    o[i]     = src[(kb + i) * stride + n];
    o[i + 8] = src[(kb + 16 + i) * stride + n];
  }
  return o;
}

// Implicit im2col fragment for conv: B[k][n] = xs[n + k]
__device__ __forceinline__ v16h frag_shift(const _Float16* xs, int n0, int k0) {
  int lane = threadIdx.x & 31;
  int n = n0 + (lane & 15);
  int kb = k0 + ((lane >> 4) << 3);
  const _Float16* p = xs + n + kb;
  v16h o;
#pragma unroll
  for (int i = 0; i < 8; i++) { o[i] = p[i]; o[i + 8] = p[i + 16]; }
  return o;
}

// Async DMA: LDS[lds_byte_addr .. +15] = MEM[gaddr .. +15], tracked by ASYNCcnt.
__device__ __forceinline__ void async_copy_b128(const _Float16* gsrc,
                                                _Float16* ldst) {
  unsigned lds_addr = (unsigned)(unsigned long long)(uintptr_t)ldst;
  unsigned long long ga = (unsigned long long)(uintptr_t)gsrc;
  asm volatile("global_load_async_to_lds_b128 %0, %1, off"
               :: "v"(lds_addr), "v"(ga) : "memory");
}
__device__ __forceinline__ void async_wait0() {
  asm volatile("s_wait_asynccnt 0x0" ::: "memory");
}

// ---------------------------------------------------------------------------
// Weight preparation (f32 -> f16, with channel padding for embeds)
// ---------------------------------------------------------------------------
__global__ void cvt16_kernel(_Float16* dst, const float* src, int n) {
  int i = blockIdx.x * blockDim.x + threadIdx.x;
  if (i < n) dst[i] = (_Float16)src[i];
}

__global__ void padcvt_kernel(_Float16* dst, const float* src, int rows,
                              int sc, int dc) {
  int i = blockIdx.x * blockDim.x + threadIdx.x;
  if (i < rows * dc) {
    int r = i / dc, c = i % dc;
    dst[i] = (c < sc) ? (_Float16)src[r * sc + c] : (_Float16)0.0f;
  }
}

__global__ void ham_kernel(float* ham) {
  int i = blockIdx.x * blockDim.x + threadIdx.x;
  if (i < 512) ham[i] = 0.54f - 0.46f * cosf(2.0f * PI_F * (float)i / 511.0f);
}

// ---------------------------------------------------------------------------
// Stage 1: conv (128 bands x 512 taps) + |.| partial pooling per 256-chunk.
// grid (128 chunks, 8 batch), block 512 (16 waves).
// Sb[b][band][c] = sum_{j in chunk c} |spec[b,band,256c+j]|
// ---------------------------------------------------------------------------
__global__ void __launch_bounds__(512, 1)
conv_pool_kernel(const float* __restrict__ x,
                 const _Float16* __restrict__ F16,
                 float* __restrict__ Sb) {
  int c = blockIdx.x, b = blockIdx.y;
  int tid = threadIdx.x;
  __shared__ alignas(16) _Float16 xs[1024];
  __shared__ float sband[128];
  if (tid < 128) sband[tid] = 0.0f;
  for (int i = tid; i < 1024; i += 512) {
    int idx = 256 * c - 256 + i;
    float v = (idx >= 0 && idx < 32768) ? x[b * 32768 + idx] : 0.0f;
    xs[i] = (_Float16)v;
  }
  __syncthreads();

  int wave = tid >> 5, lane = tid & 31;
  int m0 = (wave & 7) * 16;          // band tile
  int nbase = (wave >> 3) * 128;     // position half
  v8f acc[8] = {};
#pragma unroll 4
  for (int ks = 0; ks < 16; ks++) {
    v16h a = frag_row(F16, 512, m0, ks * 32);
#pragma unroll
    for (int nt = 0; nt < 8; nt++) {
      v16h bf = frag_shift(xs, nbase + nt * 16, ks * 32);
      acc[nt] = wmma16(a, bf, acc[nt]);
    }
  }
#pragma unroll
  for (int r = 0; r < 8; r++) {
    float s = 0.0f;
#pragma unroll
    for (int nt = 0; nt < 8; nt++) s += fabsf(acc[nt][r]);
    s += __shfl_xor(s, 8, 16);
    s += __shfl_xor(s, 4, 16);
    s += __shfl_xor(s, 2, 16);
    s += __shfl_xor(s, 1, 16);
    if ((lane & 15) == 0)
      atomicAdd(&sband[m0 + r + ((lane >> 4) << 3)], s);
  }
  __syncthreads();
  if (tid < 128) Sb[(b * 128 + tid) * 128 + c] = sband[tid];
}

// pooled[b][band][t] = (Sb[t-1] + Sb[t]) / 512
__global__ void combine_kernel(const float* __restrict__ Sb,
                               float* __restrict__ CH) {
  int i = blockIdx.x * blockDim.x + threadIdx.x;
  if (i < 8 * 128 * 128) {
    int t = i & 127;
    float v = Sb[i];
    if (t > 0) v += Sb[i - 1];
    CH[i] = v * (1.0f / 512.0f);
  }
}

// ---------------------------------------------------------------------------
// Embed: X[b][t][o] = W[o][0:192] . [pos(33) ; CH(128) ; pad] + bias[o]
// grid 8 (batch), block 256 (8 waves). K = 192, M = N = 128.
// Also emits X16[b][t][o] (f16) for downstream async B staging.
// ---------------------------------------------------------------------------
__global__ void __launch_bounds__(256, 1)
embed_kernel(const _Float16* __restrict__ W16,
             const float* __restrict__ bias,
             const float* __restrict__ CH,
             float* __restrict__ X,
             _Float16* __restrict__ X16) {
  int b = blockIdx.x, tid = threadIdx.x;
  __shared__ alignas(16) _Float16 binT[128 * 200];
  for (int i = tid; i < 128 * 192; i += 256) {
    int t = i / 192, c = i % 192;
    float v;
    float p = -1.0f + 2.0f * (float)t / 127.0f;
    if (c == 0) v = p;
    else if (c < 33) {
      int fi = (c - 1) >> 1;
      float ang = exp2f((float)fi) * p * PI_F;
      v = (((c - 1) & 1) == 0) ? sinf(ang) : cosf(ang);
    } else if (c < 161) {
      v = CH[b * 16384 + (c - 33) * 128 + t];
    } else v = 0.0f;
    binT[t * 200 + c] = (_Float16)v;
  }
  __syncthreads();

  int wave = tid >> 5, lane = tid & 31;
  int m0 = wave * 16;
  v8f acc[8] = {};
#pragma unroll
  for (int ks = 0; ks < 6; ks++) {
    v16h a = frag_row(W16, 192, m0, ks * 32);
#pragma unroll
    for (int nt = 0; nt < 8; nt++)
      acc[nt] = wmma16(a, frag_row(binT, 200, nt * 16, ks * 32), acc[nt]);
  }
#pragma unroll
  for (int nt = 0; nt < 8; nt++) {
    int ng = nt * 16 + (lane & 15);
#pragma unroll
    for (int r = 0; r < 8; r++) {
      int mg = m0 + r + ((lane >> 4) << 3);
      float v = acc[nt][r] + bias[mg];
      X[b * 16384 + ng * 128 + mg] = v;
      X16[b * 16384 + ng * 128 + mg] = (_Float16)v;
    }
  }
}

// ---------------------------------------------------------------------------
// Generic WMMA GEMM: out[b, m, n] = sum_k A16[m, k] * Bsrc[b, k, n] (+epilogue)
// grid (B, M/128), block 256 (8 waves). N = 128 fixed, K multiple of 64.
// If B16 != null (layout [b][n][K], K contiguous), the LDS B staging is done
// with GLOBAL_LOAD_ASYNC_TO_LDS_B128 (ASYNCcnt) — per-lane 16B DMA implements
// the [n][k] transpose placement with no VALU conversion. Else f32+convert.
// Optional out16: transposed f16 copy [b][n][m] for the next GEMM's B16.
// ---------------------------------------------------------------------------
__global__ void __launch_bounds__(256, 1)
gemm_kernel(const _Float16* __restrict__ A, int astr,
            const float* __restrict__ bias,
            const float* __restrict__ Bsrc, int bk, int bn, int bb,
            const _Float16* __restrict__ B16,
            float* __restrict__ out, int om, int on, int ob,
            _Float16* __restrict__ out16,
            const float* __restrict__ res,
            const float* __restrict__ scale,
            int K, int relu) {
  int b = blockIdx.x, mbase = blockIdx.y * 128;
  int tid = threadIdx.x, wave = tid >> 5, lane = tid & 31;
  __shared__ alignas(16) _Float16 BsT[128 * 72];  // 144B rows: 16B aligned
  v8f acc[8] = {};
  int nchunk = K >> 6;
  for (int kc = 0; kc < nchunk; kc++) {
    if (B16) {
      // 128 rows x 64 halves = 1024 x 16B async lane-copies (4 per thread)
      const _Float16* src = B16 + b * (128 * K) + kc * 64;
#pragma unroll
      for (int j = 0; j < 4; j++) {
        int idx = tid + j * 256;
        int n = idx >> 3, k8 = idx & 7;
        async_copy_b128(src + n * K + k8 * 8, &BsT[n * 72 + k8 * 8]);
      }
      async_wait0();
    } else {
      for (int i = tid; i < 128 * 64; i += 256) {
        int n = i >> 6, k = i & 63;
        BsT[n * 72 + k] = (_Float16)Bsrc[b * bb + (kc * 64 + k) * bk + n * bn];
      }
    }
    // Prefetch the next K-chunk of the B source (global_prefetch_b8).
    if (kc + 1 < nchunk && tid < 128)
      __builtin_prefetch(&Bsrc[b * bb + ((kc + 1) * 64) * bk + tid * bn], 0, 1);
    __syncthreads();
    int m0 = mbase + wave * 16;
#pragma unroll
    for (int ks = 0; ks < 2; ks++) {
      v16h a = frag_row(A, astr, m0, kc * 64 + ks * 32);
#pragma unroll
      for (int nt = 0; nt < 8; nt++)
        acc[nt] = wmma16(a, frag_row(BsT, 72, nt * 16, ks * 32), acc[nt]);
    }
    __syncthreads();
  }
#pragma unroll
  for (int nt = 0; nt < 8; nt++) {
    int ng = nt * 16 + (lane & 15);
#pragma unroll
    for (int r = 0; r < 8; r++) {
      int mg = mbase + wave * 16 + r + ((lane >> 4) << 3);
      float v = acc[nt][r];
      if (bias) v += bias[mg];
      if (relu) v = fmaxf(v, 0.0f);
      if (scale) v *= scale[mg];
      int idx = b * ob + mg * om + ng * on;
      if (res) v += res[idx];
      out[idx] = v;
      if (out16) out16[b * 16384 + ng * 128 + mg] = (_Float16)v;
    }
  }
}

// ---------------------------------------------------------------------------
// Fused attention per (head, batch): scores WMMA (K=32) -> shuffle softmax
// -> attn.V WMMA (K=128). QKV layout [b][384][t]; outputs AttO[b][128][t] f32
// and AttO16[b][t][128] f16 (for async B staging of the W_O GEMM).
// ---------------------------------------------------------------------------
__global__ void __launch_bounds__(256, 1)
attn_kernel(const float* __restrict__ QKV, float* __restrict__ AttO,
            _Float16* __restrict__ AttO16) {
  int h = blockIdx.x, b = blockIdx.y;
  int tid = threadIdx.x, wave = tid >> 5, lane = tid & 31;
  __shared__ alignas(16) _Float16 Qs[32 * 128];   // [d][t]
  __shared__ alignas(16) _Float16 Kt[128 * 56];   // [t][d]
  __shared__ alignas(16) _Float16 Vs[32 * 128];   // [d][t]
  __shared__ alignas(16) _Float16 As[128 * 128];  // [q_t][k_t]
  const float* base = QKV + b * 49152;
  for (int i = tid; i < 32 * 128; i += 256) {
    int d = i >> 7, t = i & 127;
    Qs[i] = (_Float16)(base[(h * 32 + d) * 128 + t] * 0.17677669529663687f);
    Kt[t * 56 + d] = (_Float16)(base[(128 + h * 32 + d) * 128 + t]);
    Vs[i] = (_Float16)(base[(256 + h * 32 + d) * 128 + t]);
  }
  __syncthreads();

  int m0 = wave * 16;
  v8f sc[8] = {};
  v16h aq = frag_col(Qs, 128, m0, 0);   // A(m=q_t, k=d) = Qs[d*128 + q_t]
#pragma unroll
  for (int nt = 0; nt < 8; nt++)
    sc[nt] = wmma16(aq, frag_row(Kt, 56, nt * 16, 0), sc[nt]);

#pragma unroll
  for (int r = 0; r < 8; r++) {
    float mx = -3.0e38f;
#pragma unroll
    for (int nt = 0; nt < 8; nt++) mx = fmaxf(mx, sc[nt][r]);
    mx = fmaxf(mx, __shfl_xor(mx, 8, 16));
    mx = fmaxf(mx, __shfl_xor(mx, 4, 16));
    mx = fmaxf(mx, __shfl_xor(mx, 2, 16));
    mx = fmaxf(mx, __shfl_xor(mx, 1, 16));
    float sm = 0.0f;
#pragma unroll
    for (int nt = 0; nt < 8; nt++) {
      float e = expf(sc[nt][r] - mx);
      sc[nt][r] = e;
      sm += e;
    }
    sm += __shfl_xor(sm, 8, 16);
    sm += __shfl_xor(sm, 4, 16);
    sm += __shfl_xor(sm, 2, 16);
    sm += __shfl_xor(sm, 1, 16);
    float inv = 1.0f / sm;
    int mg = m0 + r + ((lane >> 4) << 3);
#pragma unroll
    for (int nt = 0; nt < 8; nt++)
      As[mg * 128 + nt * 16 + (lane & 15)] = (_Float16)(sc[nt][r] * inv);
  }
  __syncthreads();

  v8f o[2] = {};
#pragma unroll
  for (int ks = 0; ks < 4; ks++) {
    v16h aa = frag_row(As, 128, m0, ks * 32);
#pragma unroll
    for (int nt = 0; nt < 2; nt++)    // B(k=k_t, n=d) = Vs[d*128 + k_t]
      o[nt] = wmma16(aa, frag_row(Vs, 128, nt * 16, ks * 32), o[nt]);
  }
#pragma unroll
  for (int nt = 0; nt < 2; nt++) {
    int ng = nt * 16 + (lane & 15);
#pragma unroll
    for (int r = 0; r < 8; r++) {
      int mg = m0 + r + ((lane >> 4) << 3);
      float v = o[nt][r];
      AttO[b * 16384 + (h * 32 + ng) * 128 + mg] = v;
      AttO16[b * 16384 + mg * 128 + h * 32 + ng] = (_Float16)v;
    }
  }
}

// LayerNorm over D=128; also emits f16 copy. grid = B*S rows, block 128.
__global__ void ln_kernel(const float* __restrict__ in,
                          const float* __restrict__ g,
                          const float* __restrict__ bb,
                          float* __restrict__ out,
                          _Float16* __restrict__ out16) {
  int row = blockIdx.x, d = threadIdx.x;
  float v = in[row * 128 + d];
  __shared__ float red[128];
  red[d] = v;
  __syncthreads();
  for (int s = 64; s > 0; s >>= 1) {
    if (d < s) red[d] += red[d + s];
    __syncthreads();
  }
  float mu = red[0] * (1.0f / 128.0f);
  __syncthreads();
  float c = v - mu;
  red[d] = c * c;
  __syncthreads();
  for (int s = 64; s > 0; s >>= 1) {
    if (d < s) red[d] += red[d + s];
    __syncthreads();
  }
  float var = red[0] * (1.0f / 128.0f);
  float r = c * rsqrtf(var + 1e-5f) * g[d] + bb[d];
  out[row * 128 + d] = r;
  out16[row * 128 + d] = (_Float16)r;
}

// env head + softmax over 128 frames + iterative top-16 + sparse scatter.
// grid 8 (batch), block 128.
__global__ void env_topk_kernel(const float* __restrict__ H,
                                const float* __restrict__ ew,
                                const float* __restrict__ eb,
                                float* __restrict__ CH) {
  int b = blockIdx.x, t = threadIdx.x;
  float s = eb[0];
  for (int d = 0; d < 128; d++) s += ew[d] * H[b * 16384 + t * 128 + d];
  __shared__ float nor[128], red[128];
  __shared__ int redi[128];
  __shared__ float vals[16];
  __shared__ int idxs[16];
  red[t] = s;
  __syncthreads();
  for (int st = 64; st > 0; st >>= 1) {
    if (t < st) red[t] = fmaxf(red[t], red[t + st]);
    __syncthreads();
  }
  float mx = red[0];
  __syncthreads();
  float e = expf(s - mx);
  red[t] = e;
  __syncthreads();
  for (int st = 64; st > 0; st >>= 1) {
    if (t < st) red[t] += red[t + st];
    __syncthreads();
  }
  float tot = red[0];
  __syncthreads();
  nor[t] = e / tot;
  __syncthreads();
  for (int a = 0; a < 16; a++) {
    red[t] = nor[t];
    redi[t] = t;
    __syncthreads();
    for (int st = 64; st > 0; st >>= 1) {
      if (t < st && red[t + st] > red[t]) { red[t] = red[t + st]; redi[t] = redi[t + st]; }
      __syncthreads();
    }
    if (t == 0) { vals[a] = red[0]; idxs[a] = redi[0]; }
    __syncthreads();
    if (t == idxs[a]) nor[t] = -1.0f;
    __syncthreads();
  }
  // thread t acts as channel d: CH[b][d][t'] sparse
  float* dst = CH + b * 16384 + t * 128;
  for (int j = 0; j < 128; j++) dst[j] = 0.0f;
  for (int a = 0; a < 16; a++) {
    int ti = idxs[a];
    dst[ti] = H[b * 16384 + ti * 128 + t] * vals[a];
  }
}

// Overlap-add: out[b][n] = sum over the <=2 frames covering n.
__global__ void ola_kernel(const float* __restrict__ FR, float* __restrict__ out) {
  int i = blockIdx.x * blockDim.x + threadIdx.x;
  if (i < 8 * 32768) {
    int b = i >> 15, n = i & 32767;
    int t0 = n >> 8;
    float acc = 0.0f;
    int tA = t0 - 1;
    if (tA >= 0) acc += FR[(b * 128 + tA) * 512 + (n - 256 * tA)];
    if (t0 < 128) acc += FR[(b * 128 + t0) * 512 + (n - 256 * t0)];
    out[i] = acc;
  }
}

// ---------------------------------------------------------------------------
// Host orchestration
// ---------------------------------------------------------------------------
#define L1D(kern, n, ...) \
  kern<<<dim3(((n) + 255) / 256), dim3(256), 0, stream>>>(__VA_ARGS__)

extern "C" void kernel_launch(void* const* d_in, const int* in_sizes, int n_in,
                              void* d_out, int out_size, void* d_ws, size_t ws_size,
                              hipStream_t stream) {
  const float* x        = (const float*)d_in[0];
  const float* filters  = (const float*)d_in[1];
  const float* embed_w  = (const float*)d_in[2];
  const float* embed_b  = (const float*)d_in[3];
  const float* embed2_w = (const float*)d_in[4];
  const float* embed2_b = (const float*)d_in[5];
  const float* env_w    = (const float*)d_in[6];
  const float* env_b    = (const float*)d_in[7];
  const float* samp_w   = (const float*)d_in[8];
  const float* samp_b   = (const float*)d_in[9];

  size_t off = 0;
  auto a16 = [&](size_t n) {
    _Float16* p = (_Float16*)((char*)d_ws + off);
    off += ((n * sizeof(_Float16) + 255) & ~(size_t)255);
    return p;
  };
  auto af = [&](size_t n) {
    float* p = (float*)((char*)d_ws + off);
    off += ((n * sizeof(float) + 255) & ~(size_t)255);
    return p;
  };

  _Float16* F16   = a16(128 * 512);
  _Float16* E1w16 = a16(128 * 192);
  _Float16* E2w16 = a16(128 * 192);
  _Float16* SW16  = a16(512 * 128);
  _Float16* LW16  = a16(12 * 98304);  // per layer: qkv(49152) wo w1 w2 (16384 each)
  _Float16* X16   = a16(8 * 128 * 128);   // [b][t][d]
  _Float16* AO16  = a16(8 * 128 * 128);   // [b][t][d]
  _Float16* H116  = a16(8 * 128 * 128);   // [b][t][f]
  float* ham  = af(512);
  float* Sb   = af(8 * 128 * 128);
  float* CH   = af(8 * 128 * 128);
  float* X    = af(8 * 128 * 128);
  float* QKV  = af(8 * 384 * 128);    // also reused as FFN hidden H1 (f32)
  float* AttO = af(8 * 128 * 128);
  float* TMP  = af(8 * 128 * 128);
  float* FR   = af(8 * 128 * 512);

  // ---- weight prep ----
  L1D(cvt16_kernel, 128 * 512, F16, filters, 128 * 512);
  L1D(padcvt_kernel, 128 * 192, E1w16, embed_w, 128, 161, 192);
  L1D(padcvt_kernel, 128 * 192, E2w16, embed2_w, 128, 161, 192);
  L1D(cvt16_kernel, 512 * 128, SW16, samp_w, 512 * 128);
  L1D(ham_kernel, 512, ham);
  for (int l = 0; l < 12; l++) {
    const float* wq = (const float*)d_in[10 + l * 12 + 0];
    const float* wo = (const float*)d_in[10 + l * 12 + 2];
    const float* w1 = (const float*)d_in[10 + l * 12 + 4];
    const float* w2 = (const float*)d_in[10 + l * 12 + 6];
    _Float16* base = LW16 + l * 98304;
    L1D(cvt16_kernel, 384 * 128, base, wq, 384 * 128);
    L1D(cvt16_kernel, 128 * 128, base + 49152, wo, 128 * 128);
    L1D(cvt16_kernel, 128 * 128, base + 65536, w1, 128 * 128);
    L1D(cvt16_kernel, 128 * 128, base + 81920, w2, 128 * 128);
  }

  // ---- conv + pooling ----
  conv_pool_kernel<<<dim3(128, 8), dim3(512), 0, stream>>>(x, F16, Sb);
  L1D(combine_kernel, 8 * 128 * 128, Sb, CH);

  // ---- embed 1 ----
  embed_kernel<<<dim3(8), dim3(256), 0, stream>>>(E1w16, embed_b, CH, X, X16);

  auto run_layer = [&](int l) {
    const _Float16* Wq = LW16 + l * 98304;
    const _Float16* Wo = Wq + 49152;
    const _Float16* W1 = Wq + 65536;
    const _Float16* W2 = Wq + 81920;
    const float* bq  = (const float*)d_in[10 + l * 12 + 1];
    const float* bo  = (const float*)d_in[10 + l * 12 + 3];
    const float* b1  = (const float*)d_in[10 + l * 12 + 5];
    const float* b2  = (const float*)d_in[10 + l * 12 + 7];
    const float* g1  = (const float*)d_in[10 + l * 12 + 8];
    const float* be1 = (const float*)d_in[10 + l * 12 + 9];
    const float* g2  = (const float*)d_in[10 + l * 12 + 10];
    const float* be2 = (const float*)d_in[10 + l * 12 + 11];
    // QKV[b][o][t] = wqkv . X^T + b   (B staged async from X16)
    gemm_kernel<<<dim3(8, 3), dim3(256), 0, stream>>>(
        Wq, 128, bq, X, 1, 128, 16384, X16, QKV, 128, 1, 49152,
        nullptr, nullptr, nullptr, 128, 0);
    attn_kernel<<<dim3(4, 8), dim3(256), 0, stream>>>(QKV, AttO, AO16);
    // TMP[b][t][o] = wo . AttO + bo + X (residual); B staged async from AO16
    gemm_kernel<<<dim3(8, 1), dim3(256), 0, stream>>>(
        Wo, 128, bo, AttO, 128, 1, 16384, AO16, TMP, 1, 128, 16384,
        nullptr, X, nullptr, 128, 0);
    ln_kernel<<<dim3(8 * 128), dim3(128), 0, stream>>>(TMP, g1, be1, X, X16);
    // H1[b][f][t] = relu(w1 . X^T + b1); f32 in QKV scratch, f16 copy to H116
    gemm_kernel<<<dim3(8, 1), dim3(256), 0, stream>>>(
        W1, 128, b1, X, 1, 128, 16384, X16, QKV, 128, 1, 16384,
        H116, nullptr, nullptr, 128, 1);
    // TMP[b][t][o] = w2 . H1 + b2 + X (residual); B staged async from H116
    gemm_kernel<<<dim3(8, 1), dim3(256), 0, stream>>>(
        W2, 128, b2, QKV, 128, 1, 16384, H116, TMP, 1, 128, 16384,
        nullptr, X, nullptr, 128, 0);
    ln_kernel<<<dim3(8 * 128), dim3(128), 0, stream>>>(TMP, g2, be2, X, X16);
  };

  for (int l = 0; l < 6; l++) run_layer(l);          // context encoder

  // ---- env head, softmax, top-16, sparse gating ----
  env_topk_kernel<<<dim3(8), dim3(128), 0, stream>>>(X, env_w, env_b, CH);

  // ---- embed 2 + decoder ----
  embed_kernel<<<dim3(8), dim3(256), 0, stream>>>(E2w16, embed2_b, CH, X, X16);
  for (int l = 6; l < 12; l++) run_layer(l);

  // ---- synthesis GEMM with Hamming window (B staged async from X16) ----
  gemm_kernel<<<dim3(8, 4), dim3(256), 0, stream>>>(
      SW16, 128, samp_b, X, 1, 128, 16384, X16, FR, 1, 512, 65536,
      nullptr, nullptr, ham, 128, 0);

  // ---- overlap-add ----
  L1D(ola_kernel, 8 * 32768, FR, (float*)d_out);

  (void)in_sizes; (void)n_in; (void)out_size; (void)ws_size;
}
